// RN_31104153157914
// MI455X (gfx1250) — matrix operand order repaired and checked by reading, not compiled, not run on hardware
//
#include <hip/hip_runtime.h>
#include <hip/hip_bf16.h>
#include <cstdint>

// ---------------- types ----------------
typedef __attribute__((ext_vector_type(16))) __bf16 bf16x16;
typedef __attribute__((ext_vector_type(8)))  float  floatx8;
typedef __attribute__((ext_vector_type(4))) unsigned int uint4v;

union FragU {
    uint4v q[2];
    bf16x16 v;
    unsigned short s[16];
};

// gfx1250 async global->LDS path (device pass only; falls back to reg staging)
#if defined(__gfx1250__) && __has_builtin(__builtin_amdgcn_global_load_async_to_lds_b128)
#define RN_HAS_ASYNC 1
// builtin expects: (int4(vector_size) addrspace(1)*, int4 addrspace(3)*, imm, imm)
typedef int rn_i4 __attribute__((vector_size(16)));
typedef __attribute__((address_space(1))) rn_i4* rn_gp;
typedef __attribute__((address_space(3))) rn_i4* rn_lp;
#else
#define RN_HAS_ASYNC 0
#endif

static __device__ __forceinline__ void rn_wait_async() {
#if RN_HAS_ASYNC
#if __has_builtin(__builtin_amdgcn_s_wait_asynccnt)
    __builtin_amdgcn_s_wait_asynccnt(0);
#else
    asm volatile("s_wait_asynccnt 0x0" ::: "memory");
#endif
#endif
}

static __device__ __forceinline__ unsigned short f32_to_bf16(float f) {
    unsigned int x = __float_as_uint(f);
    unsigned int r = (x + 0x7FFFu + ((x >> 16) & 1u)) >> 16;  // RNE
    return (unsigned short)r;
}
static __device__ __forceinline__ float bf16_to_f32(unsigned short u) {
    return __uint_as_float(((unsigned int)u) << 16);
}

// ---------------- conv (stride 2, pad 1, 3x3) + bias + relu ----------------
__global__ void rn_conv_relu(const float* __restrict__ x, const float* __restrict__ w,
                             const float* __restrict__ bias, float* __restrict__ y,
                             int Cin, int Cout, int Hin, int Win, int Hout, int Wout) {
    long long idx = (long long)blockIdx.x * blockDim.x + threadIdx.x;
    long long total = (long long)64 * Cout * Hout * Wout;
    if (idx >= total) return;
    int wo = idx % Wout;
    int ho = (idx / Wout) % Hout;
    int co = (idx / ((long long)Wout * Hout)) % Cout;
    int b  = idx / ((long long)Wout * Hout * Cout);
    float acc = bias[co];
    for (int ci = 0; ci < Cin; ++ci) {
        const float* xp = x + ((long long)(b * Cin + ci) * Hin) * Win;
        const float* wp = w + ((long long)(co * Cin + ci) * 9);
        #pragma unroll
        for (int kh = 0; kh < 3; ++kh) {
            int ih = ho * 2 - 1 + kh;
            if (ih < 0 || ih >= Hin) continue;
            #pragma unroll
            for (int kw = 0; kw < 3; ++kw) {
                int iw = wo * 2 - 1 + kw;
                if (iw < 0 || iw >= Win) continue;
                acc += xp[(long long)ih * Win + iw] * wp[kh * 3 + kw];
            }
        }
    }
    y[idx] = acc > 0.0f ? acc : 0.0f;
}

// ---------------- batchnorm stats (mean/var over N,H,W per channel) ----------------
__global__ void rn_bn_stats(const float* __restrict__ x, float* __restrict__ mean,
                            float* __restrict__ var, int C, int HW) {
    int c = blockIdx.x;
    __shared__ float s1[256];
    __shared__ float s2[256];
    float a = 0.0f, b = 0.0f;
    int tot = 64 * HW;
    for (int i = threadIdx.x; i < tot; i += 256) {
        int n = i / HW, hw = i % HW;
        float v = x[((long long)(n * C + c)) * HW + hw];
        a += v; b += v * v;
    }
    s1[threadIdx.x] = a; s2[threadIdx.x] = b;
    __syncthreads();
    for (int s = 128; s > 0; s >>= 1) {
        if (threadIdx.x < s) { s1[threadIdx.x] += s1[threadIdx.x + s]; s2[threadIdx.x] += s2[threadIdx.x + s]; }
        __syncthreads();
    }
    if (threadIdx.x == 0) {
        float m = s1[0] / (float)tot;
        mean[c] = m;
        var[c] = s2[0] / (float)tot - m * m;
    }
}

__global__ void rn_bn_apply(float* __restrict__ x, const float* __restrict__ g,
                            const float* __restrict__ bb, const float* __restrict__ mean,
                            const float* __restrict__ var, int C, int HW) {
    long long idx = (long long)blockIdx.x * blockDim.x + threadIdx.x;
    long long total = (long long)64 * C * HW;
    if (idx >= total) return;
    int c = (int)((idx / HW) % C);
    x[idx] = g[c] * (x[idx] - mean[c]) * rsqrtf(var[c] + 1e-5f) + bb[c];
}

// ---------------- build x_flat (64,25,258) ----------------
__global__ void rn_build_xflat(const float* __restrict__ x4, float* __restrict__ xf) {
    int idx = blockIdx.x * blockDim.x + threadIdx.x;
    int total = 64 * 25 * 258;
    if (idx >= total) return;
    int c = idx % 258;
    int p = (idx / 258) % 25;
    int b = idx / (258 * 25);
    float v;
    if (c < 256)      v = x4[((long long)(b * 256 + c)) * 25 + p];
    else if (c == 256) v = ((float)p / 5.0f - 2.0f) * 0.5f;   // true division per reference
    else               v = ((float)(p % 5) - 2.0f) * 0.5f;
    xf[idx] = v;
}

// ---------------- naive f32 GEMM (small matrices) ----------------
__global__ void rn_naive_gemm(const float* __restrict__ A, const float* __restrict__ B,
                              const float* __restrict__ bias, float* __restrict__ out,
                              int M, int K, int N, int ldb, int relu) {
    long long idx = (long long)blockIdx.x * blockDim.x + threadIdx.x;
    if (idx >= (long long)M * N) return;
    int c = (int)(idx % N);
    int r = (int)(idx / N);
    float acc = bias ? bias[c] : 0.0f;
    const float* a = A + (long long)r * K;
    const float* b = B + c;
    for (int k = 0; k < K; ++k) acc += a[k] * b[(long long)k * ldb];
    if (relu && acc < 0.0f) acc = 0.0f;
    out[idx] = acc;
}

// ---------------- build h = relu(bb[p1] + a[p2] + q) -> bf16 ----------------
__global__ void rn_build_h(const float* __restrict__ a, const float* __restrict__ bb,
                           const float* __restrict__ q, unsigned short* __restrict__ h) {
    long long idx = (long long)blockIdx.x * blockDim.x + threadIdx.x;
    long long total = (long long)64 * 625 * 2000;
    if (idx >= total) return;
    int e  = (int)(idx % 2000);
    int p  = (int)((idx / 2000) % 625);
    int b  = (int)(idx / (2000LL * 625));
    int p1 = p / 25, p2 = p % 25;
    float v = bb[((long long)(b * 25 + p1)) * 2000 + e]
            + a [((long long)(b * 25 + p2)) * 2000 + e]
            + q [(long long)b * 2000 + e];
    if (v < 0.0f) v = 0.0f;
    h[idx] = f32_to_bf16(v);
}

// ---------------- f32 -> bf16 weight conversion ----------------
__global__ void rn_to_bf16(const float* __restrict__ in, unsigned short* __restrict__ out, int n) {
    int idx = blockIdx.x * blockDim.x + threadIdx.x;
    if (idx < n) out[idx] = f32_to_bf16(in[idx]);
}

// ---------------- big WMMA GEMM: C = relu(A@B + bias) (bf16 in, bf16 out) ----------------
// Block tile 256x128, 8 waves as 4(M)x2(N), each wave owns a 64x64 tile = 4x4 WMMA frags.
// LDS rows padded to 40 bf16 (80B) so 16-lane fragment reads hit 16 distinct banks.
#define BM 256
#define BN 128
#define BK 32
#define AST 40   // padded LDS row stride (bf16 elems) for A tile
#define BST 40   // padded LDS row stride (bf16 elems) for B tile

__global__ __launch_bounds__(256) void rn_gemm_wmma(
    const unsigned short* __restrict__ A,   // M x K bf16 (row major)
    const unsigned short* __restrict__ Bw,  // K x N bf16 (row major)
    const float* __restrict__ bias,         // N
    unsigned short* __restrict__ C,         // M x N bf16, relu applied
    int M, int N, int K) {
    __shared__ __align__(16) unsigned short As[2][BM * AST];  // [m][k], padded
    __shared__ __align__(16) unsigned short Bs[2][BN * BST];  // [n][k], padded (transposed)

    const int t = threadIdx.x;
    const int lane = t & 31;
    const int wv = t >> 5;
    const int waveM = wv & 3;    // 0..3 -> 64-row strip
    const int waveN = wv >> 2;   // 0..1 -> 64-col strip

    const int mBase = (int)blockIdx.x * BM;
    const int nBase = (int)blockIdx.y * BN;

    // loader coordinates: A -> thread t owns full row t (32 bf16); B -> 16 bf16 per thread
    const int bKr    = t >> 3;          // 0..31 within BK
    const int bChunk = (t & 7) * 16;    // 0..112 within BN

    floatx8 acc[4][4] = {};
    uint4v bReg[2];
    const uint4v zero4 = {0u, 0u, 0u, 0u};

#if RN_HAS_ASYNC
    // A tile via GLOBAL_LOAD_ASYNC_TO_LDS_B128: layout-preserving [m][k] copy.
    // OOB handling by clamping (reads valid finite data): row-OOB rows are
    // discarded by the epilogue guard; k-tail garbage multiplies zeroed B rows.
    auto loadA = [&](int k0, int buf) {
        int gm = mBase + t;  if (gm >= M) gm = M - 1;
        #pragma unroll
        for (int cch = 0; cch < 2; ++cch) {
            int gk = k0 + cch * 16;  if (gk + 16 > K) gk = K - 16;   // K % 16 == 0
            const unsigned short* gp = A + (size_t)gm * K + gk;
            unsigned short* lp = &As[buf][t * AST + cch * 16];
            __builtin_amdgcn_global_load_async_to_lds_b128((rn_gp)gp, (rn_lp)lp, 0, 0);
            __builtin_amdgcn_global_load_async_to_lds_b128((rn_gp)gp, (rn_lp)lp, 16, 0);
        }
    };
#else
    uint4v aReg[4];
    auto gloadA = [&](int k0) {
        int gm = mBase + t;
        #pragma unroll
        for (int cch = 0; cch < 2; ++cch) {
            int gk = k0 + cch * 16;
            if (gm < M && gk < K) {
                const uint4v* p = (const uint4v*)(A + (size_t)gm * K + gk);
                aReg[2 * cch] = p[0]; aReg[2 * cch + 1] = p[1];
            } else { aReg[2 * cch] = zero4; aReg[2 * cch + 1] = zero4; }
        }
    };
    auto sstoreA = [&](int buf) {
        #pragma unroll
        for (int cch = 0; cch < 2; ++cch) {
            uint4v* d = (uint4v*)&As[buf][t * AST + cch * 16];
            d[0] = aReg[2 * cch]; d[1] = aReg[2 * cch + 1];
        }
    };
#endif
    auto gloadB = [&](int k0) {
        int bk = k0 + bKr;
        int gn = nBase + bChunk;
        if (bk < K && gn < N) {
            const uint4v* p = (const uint4v*)(Bw + (size_t)bk * N + gn);
            bReg[0] = p[0]; bReg[1] = p[1];
        } else { bReg[0] = zero4; bReg[1] = zero4; }
    };
    auto sstoreB = [&](int buf) {
        FragU u; u.q[0] = bReg[0]; u.q[1] = bReg[1];
        unsigned short* bsb = &Bs[buf][0];
        #pragma unroll
        for (int j = 0; j < 16; ++j)
            bsb[(bChunk + j) * BST + bKr] = u.s[j];   // transpose into [n][k]
    };

    const int KT = (K + BK - 1) / BK;
#if RN_HAS_ASYNC
    loadA(0, 0);
    gloadB(0);
    sstoreB(0);
    rn_wait_async();
#else
    gloadA(0);
    gloadB(0);
    sstoreA(0);
    sstoreB(0);
#endif
    __syncthreads();

    const int mrow = lane & 15;
    const int koff = (lane >> 4) * 8;    // A frag: K runs {koff..koff+7, koff+16..koff+23}
    const int kb   = (lane >> 4) * 8;    // B frag: K runs {kb..kb+7, kb+16..kb+23}

    for (int kt = 0; kt < KT; ++kt) {
        int cb = kt & 1;
        if (kt + 1 < KT) {
#if RN_HAS_ASYNC
            loadA((kt + 1) * BK, cb ^ 1);    // DMA directly into other LDS buffer
#else
            gloadA((kt + 1) * BK);
#endif
            gloadB((kt + 1) * BK);           // overlap global loads with WMMA
        }

        FragU af[4];
        #pragma unroll
        for (int fm = 0; fm < 4; ++fm) {
            const unsigned short* ap = &As[cb][(waveM * 64 + fm * 16 + mrow) * AST + koff];
            af[fm].q[0] = *(const uint4v*)ap;          // K koff..koff+7
            af[fm].q[1] = *(const uint4v*)(ap + 16);   // K koff+16..koff+23
        }
        #pragma unroll
        for (int fn = 0; fn < 4; ++fn) {
            FragU bf;
            const unsigned short* bp = &Bs[cb][(waveN * 64 + fn * 16 + mrow) * BST + kb];
            bf.q[0] = *(const uint4v*)bp;        // K kb..kb+7
            bf.q[1] = *(const uint4v*)(bp + 16); // K kb+16..kb+23
            #pragma unroll
            for (int fm = 0; fm < 4; ++fm)
                acc[fm][fn] = __builtin_amdgcn_wmma_f32_16x16x32_bf16(
                    false, af[fm].v, false, bf.v, (short)0, acc[fm][fn], false, false);
        }
        if (kt + 1 < KT) {
#if !RN_HAS_ASYNC
            sstoreA(cb ^ 1);
#endif
            sstoreB(cb ^ 1);
        }
        rn_wait_async();
        __syncthreads();
    }

    // epilogue: bias + relu + bf16 store
    const int nlane = lane & 15;
    const int mhi = (lane >> 4) * 8;
    #pragma unroll
    for (int fm = 0; fm < 4; ++fm) {
        #pragma unroll
        for (int fn = 0; fn < 4; ++fn) {
            int gnc = nBase + waveN * 64 + fn * 16 + nlane;
            #pragma unroll
            for (int r = 0; r < 8; ++r) {
                int gm = mBase + waveM * 64 + fm * 16 + mhi + r;
                if (gm < M && gnc < N) {
                    float v = acc[fm][fn][r] + bias[gnc];
                    v = v > 0.0f ? v : 0.0f;
                    C[(size_t)gm * N + gnc] = f32_to_bf16(v);
                }
            }
        }
    }
}

// ---------------- sum over 625 positions ----------------
__global__ void rn_sum_p(const unsigned short* __restrict__ H, float* __restrict__ xg) {
    int idx = blockIdx.x * blockDim.x + threadIdx.x;
    if (idx >= 64 * 2000) return;
    int e = idx % 2000, b = idx / 2000;
    const unsigned short* p = H + (size_t)b * 625 * 2000 + e;
    float s = 0.0f;
    for (int i = 0; i < 625; ++i) s += bf16_to_f32(p[(size_t)i * 2000]);
    xg[idx] = s;
}

// ---------------- log_softmax on (64,10) ----------------
__global__ void rn_logsoftmax(const float* __restrict__ y, float* __restrict__ out) {
    int b = threadIdx.x;
    if (b >= 64) return;
    const float* r = y + b * 10;
    float m = r[0];
    for (int i = 1; i < 10; ++i) m = fmaxf(m, r[i]);
    float s = 0.0f;
    for (int i = 0; i < 10; ++i) s += expf(r[i] - m);
    float ls = logf(s);
    for (int i = 0; i < 10; ++i) out[b * 10 + i] = r[i] - m - ls;
}

// ---------------- host launcher ----------------
extern "C" void kernel_launch(void* const* d_in, const int* in_sizes, int n_in,
                              void* d_out, int out_size, void* d_ws, size_t ws_size,
                              hipStream_t stream) {
    const float* img   = (const float*)d_in[0];
    const float* qst   = (const float*)d_in[1];
    const float* c1_w  = (const float*)d_in[2];
    const float* c1_b  = (const float*)d_in[3];
    const float* bn1_g = (const float*)d_in[4];
    const float* bn1_b = (const float*)d_in[5];
    const float* c2_w  = (const float*)d_in[6];
    const float* c2_b  = (const float*)d_in[7];
    const float* bn2_g = (const float*)d_in[8];
    const float* bn2_b = (const float*)d_in[9];
    const float* c3_w  = (const float*)d_in[10];
    const float* c3_b  = (const float*)d_in[11];
    const float* bn3_g = (const float*)d_in[12];
    const float* bn3_b = (const float*)d_in[13];
    const float* c4_w  = (const float*)d_in[14];
    const float* c4_b  = (const float*)d_in[15];
    const float* bn4_g = (const float*)d_in[16];
    const float* bn4_b = (const float*)d_in[17];
    const float* g1_w  = (const float*)d_in[18];
    const float* g1_b  = (const float*)d_in[19];
    const float* g2_w  = (const float*)d_in[20];
    const float* g2_b  = (const float*)d_in[21];
    const float* g3_w  = (const float*)d_in[22];
    const float* g3_b  = (const float*)d_in[23];
    const float* g4_w  = (const float*)d_in[24];
    const float* g4_b  = (const float*)d_in[25];
    const float* f1_w  = (const float*)d_in[26];
    const float* f1_b  = (const float*)d_in[27];
    const float* fc2_w = (const float*)d_in[28];
    const float* fc2_b = (const float*)d_in[29];
    const float* fc3_w = (const float*)d_in[30];
    const float* fc3_b = (const float*)d_in[31];
    const float* fc4_w = (const float*)d_in[32];
    const float* fc4_b = (const float*)d_in[33];
    (void)n_in; (void)in_sizes; (void)out_size; (void)ws_size;

    char* ws = (char*)d_ws;
    size_t off = 0;
    auto walloc = [&](size_t bytes) -> void* {
        void* p = ws + off;
        off += (bytes + 255) & ~(size_t)255;
        return p;
    };

    float* c1   = (float*)walloc((size_t)64 * 32 * 38 * 38 * 4);
    float* c2   = (float*)walloc((size_t)64 * 64 * 19 * 19 * 4);
    float* c3   = (float*)walloc((size_t)64 * 128 * 10 * 10 * 4);
    float* c4   = (float*)walloc((size_t)64 * 256 * 5 * 5 * 4);
    float* mean = (float*)walloc(256 * 4);
    float* var  = (float*)walloc(256 * 4);
    float* xfl  = (float*)walloc((size_t)64 * 25 * 258 * 4);
    float* amat = (float*)walloc((size_t)64 * 25 * 2000 * 4);
    float* bmat = (float*)walloc((size_t)64 * 25 * 2000 * 4);
    float* qmat = (float*)walloc((size_t)64 * 2000 * 4);
    unsigned short* w2 = (unsigned short*)walloc((size_t)2000 * 2000 * 2);
    unsigned short* w3 = (unsigned short*)walloc((size_t)2000 * 2000 * 2);
    unsigned short* w4 = (unsigned short*)walloc((size_t)2000 * 2000 * 2);
    unsigned short* H0 = (unsigned short*)walloc((size_t)40000 * 2000 * 2);
    unsigned short* H1 = (unsigned short*)walloc((size_t)40000 * 2000 * 2);
    float* xg = (float*)walloc((size_t)64 * 2000 * 4);
    float* xf = (float*)walloc((size_t)64 * 1000 * 4);
    float* y2 = (float*)walloc((size_t)64 * 500 * 4);
    float* y3 = (float*)walloc((size_t)64 * 100 * 4);
    float* y4 = (float*)walloc((size_t)64 * 10 * 4);

    auto blocks = [](long long total) { return (unsigned)((total + 255) / 256); };

    // conv stack: conv+relu, bn stats, bn apply (in place)
    rn_conv_relu<<<blocks((long long)64 * 32 * 38 * 38), 256, 0, stream>>>(img, c1_w, c1_b, c1, 3, 32, 75, 75, 38, 38);
    rn_bn_stats<<<32, 256, 0, stream>>>(c1, mean, var, 32, 38 * 38);
    rn_bn_apply<<<blocks((long long)64 * 32 * 38 * 38), 256, 0, stream>>>(c1, bn1_g, bn1_b, mean, var, 32, 38 * 38);

    rn_conv_relu<<<blocks((long long)64 * 64 * 19 * 19), 256, 0, stream>>>(c1, c2_w, c2_b, c2, 32, 64, 38, 38, 19, 19);
    rn_bn_stats<<<64, 256, 0, stream>>>(c2, mean, var, 64, 19 * 19);
    rn_bn_apply<<<blocks((long long)64 * 64 * 19 * 19), 256, 0, stream>>>(c2, bn2_g, bn2_b, mean, var, 64, 19 * 19);

    rn_conv_relu<<<blocks((long long)64 * 128 * 10 * 10), 256, 0, stream>>>(c2, c3_w, c3_b, c3, 64, 128, 19, 19, 10, 10);
    rn_bn_stats<<<128, 256, 0, stream>>>(c3, mean, var, 128, 100);
    rn_bn_apply<<<blocks((long long)64 * 128 * 100), 256, 0, stream>>>(c3, bn3_g, bn3_b, mean, var, 128, 100);

    rn_conv_relu<<<blocks((long long)64 * 256 * 5 * 5), 256, 0, stream>>>(c3, c4_w, c4_b, c4, 128, 256, 10, 10, 5, 5);
    rn_bn_stats<<<256, 256, 0, stream>>>(c4, mean, var, 256, 25);
    rn_bn_apply<<<blocks((long long)64 * 256 * 25), 256, 0, stream>>>(c4, bn4_g, bn4_b, mean, var, 256, 25);

    // x_flat and g1 projections
    rn_build_xflat<<<blocks(64 * 25 * 258), 256, 0, stream>>>(c4, xfl);
    rn_naive_gemm<<<blocks((long long)1600 * 2000), 256, 0, stream>>>(xfl, g1_w,              nullptr, amat, 1600, 258, 2000, 2000, 0);
    rn_naive_gemm<<<blocks((long long)1600 * 2000), 256, 0, stream>>>(xfl, g1_w + 258 * 2000, nullptr, bmat, 1600, 258, 2000, 2000, 0);
    rn_naive_gemm<<<blocks((long long)64 * 2000), 256, 0, stream>>>(qst, g1_w + 516 * 2000,   g1_b,    qmat, 64, 11, 2000, 2000, 0);

    // h tensor (bf16)
    rn_build_h<<<blocks((long long)64 * 625 * 2000), 256, 0, stream>>>(amat, bmat, qmat, H0);

    // bf16 weights
    rn_to_bf16<<<blocks((long long)2000 * 2000), 256, 0, stream>>>(g2_w, w2, 2000 * 2000);
    rn_to_bf16<<<blocks((long long)2000 * 2000), 256, 0, stream>>>(g3_w, w3, 2000 * 2000);
    rn_to_bf16<<<blocks((long long)2000 * 2000), 256, 0, stream>>>(g4_w, w4, 2000 * 2000);

    // big WMMA GEMMs: M=40000, N=K=2000
    dim3 ggrid((40000 + BM - 1) / BM, (2000 + BN - 1) / BN);
    rn_gemm_wmma<<<ggrid, 256, 0, stream>>>(H0, w2, g2_b, H1, 40000, 2000, 2000);
    rn_gemm_wmma<<<ggrid, 256, 0, stream>>>(H1, w3, g3_b, H0, 40000, 2000, 2000);
    rn_gemm_wmma<<<ggrid, 256, 0, stream>>>(H0, w4, g4_b, H1, 40000, 2000, 2000);

    // sum over positions, then MLP head
    rn_sum_p<<<blocks(64 * 2000), 256, 0, stream>>>(H1, xg);
    rn_naive_gemm<<<blocks((long long)64 * 1000), 256, 0, stream>>>(xg, f1_w,  f1_b,  xf, 64, 2000, 1000, 1000, 1);
    rn_naive_gemm<<<blocks((long long)64 * 500), 256, 0, stream>>>(xf, fc2_w, fc2_b, y2, 64, 1000, 500, 500, 1);
    rn_naive_gemm<<<blocks((long long)64 * 100), 256, 0, stream>>>(y2, fc3_w, fc3_b, y3, 64, 500, 100, 100, 1);
    rn_naive_gemm<<<blocks((long long)64 * 10), 256, 0, stream>>>(y3, fc4_w, fc4_b, y4, 64, 100, 10, 10, 0);
    rn_logsoftmax<<<1, 64, 0, stream>>>(y4, (float*)d_out);
}
